// Model_32633161515217
// MI455X (gfx1250) — compile-verified, matrix operand order checked by reading, not run on hardware
//
#include <hip/hip_runtime.h>
#include <hip/hip_bf16.h>
#include <math.h>

typedef __bf16 bf16_t;
typedef __attribute__((ext_vector_type(16))) __bf16 v16bf;
typedef __attribute__((ext_vector_type(8)))  float  v8f;

namespace cfg {
constexpr int B = 32, C = 321, L = 720, SEG = 48, S = 15, D = 512, G = 1536;
constexpr int R = B * C;                      // 10272 sequences (= 32 * 321)
constexpr long long RS = (long long)R * S;    // 154080 decoder rows (= 32 * 4815)
constexpr int MP = 4832;                      // C*S = 4815 padded to multiple of 32
}

union FragU { v16bf v; int4 q[2]; };

// Load one 16x32 bf16 fragment (A or B^T operand) from row-major [16 x ld] tile.
// ISA layout (CDNA5 ISA, 16-bit A 16x32): lane<16 -> row=lane, K 0..7 & 16..23;
// lane>=16 -> row=lane-16, K 8..15 & 24..31.  Two b128 loads per lane.
__device__ __forceinline__ v16bf load_frag(const bf16_t* __restrict__ base, int ld) {
  const int lane = threadIdx.x & 31;
  const int row  = lane & 15;
  const int ko   = (lane >> 4) << 3;   // 0 or 8
  const bf16_t* p = base + (size_t)row * ld + ko;
  FragU f;
  f.q[0] = *(const int4*)(p);
  f.q[1] = *(const int4*)(p + 16);
  return f.v;
}

__device__ __forceinline__ v8f wmma_bf16(v16bf a, v16bf b, v8f c) {
  return __builtin_amdgcn_wmma_f32_16x16x32_bf16(false, a, false, b, (short)0, c,
                                                 false, false);
}

__device__ __forceinline__ float sigmoidf_(float x) {
  return 1.f / (1.f + __expf(-x));
}

// -------- prep: convert weights to bf16, zero hidden state -----------------
__global__ __launch_bounds__(256) void prep_kernel(
    const float* __restrict__ W_ih, const float* __restrict__ W_hh,
    const float* __restrict__ W_pred,
    bf16_t* __restrict__ W_ih_bf, bf16_t* __restrict__ W_hh_bf,
    bf16_t* __restrict__ W_pred_bf,
    float* __restrict__ h, bf16_t* __restrict__ h_bf) {
  const long long i = (long long)blockIdx.x * blockDim.x + threadIdx.x;
  if (i < (long long)cfg::G * cfg::D) {
    W_ih_bf[i] = (bf16_t)W_ih[i];
    W_hh_bf[i] = (bf16_t)W_hh[i];
  }
  if (i < (long long)cfg::SEG * cfg::D) W_pred_bf[i] = (bf16_t)W_pred[i];
  if (i < (long long)cfg::R * cfg::D) { h[i] = 0.f; h_bf[i] = (bf16_t)0.f; }
}

// -------- segment embedding: relu((x - last) @ W_emb^T + b_emb) -> bf16 ----
__global__ __launch_bounds__(256) void embed_kernel(
    const float* __restrict__ x_enc, const float* __restrict__ W_emb,
    const float* __restrict__ b_emb, bf16_t* __restrict__ X_emb) {
  const long long rr = blockIdx.x;          // (r, s) row in [0, R*S)
  const int s = (int)(rr % cfg::S);
  const long long r = rr / cfg::S;
  const int b = (int)(r / cfg::C);
  const int c = (int)(r % cfg::C);
  __shared__ float xv[cfg::SEG];
  if (threadIdx.x < cfg::SEG) {
    const float last = x_enc[((size_t)b * cfg::L + (cfg::L - 1)) * cfg::C + c];
    xv[threadIdx.x] =
        x_enc[((size_t)b * cfg::L + (s * cfg::SEG + threadIdx.x)) * cfg::C + c] - last;
  }
  __syncthreads();
  for (int d = threadIdx.x; d < cfg::D; d += blockDim.x) {
    float acc = b_emb[d];
    const float* w = W_emb + (size_t)d * cfg::SEG;
#pragma unroll
    for (int t = 0; t < cfg::SEG; ++t) acc = fmaf(xv[t], w[t], acc);
    acc = acc > 0.f ? acc : 0.f;
    X_emb[rr * cfg::D + d] = (bf16_t)acc;
  }
}

// -------- generic WMMA GEMM: C[M,N] = A[M,K](bf16, lda) @ W[N,K]^T + bias --
// wave tile 32(M) x 64(N): 2 A frags x 4 B frags -> 8 WMMA per K=32 step
// (~22 FLOP per L2 byte vs ~11 for a 16x64 tile).  Block = 8 waves.
// M must be a multiple of 32, N a multiple of 64, K a multiple of 32.
__global__ __launch_bounds__(256) void gemm_bf16_bias(
    const bf16_t* __restrict__ A, int lda, const bf16_t* __restrict__ W,
    const float* __restrict__ bias, float* __restrict__ Cmat,
    int M, int N, int K) {
  const int wave = threadIdx.x >> 5;
  const int lane = threadIdx.x & 31;
  const int tilesN = N >> 6;
  const long long nt = (long long)(M >> 5) * tilesN;
  const long long tid = (long long)blockIdx.x * (blockDim.x >> 5) + wave;
  if (tid >= nt) return;                 // wave-uniform: EXEC all-ones for WMMA
  const int tn = (int)(tid % tilesN);
  const long long tm = tid / tilesN;

  const bf16_t* Ab = A + (size_t)tm * 32 * lda;
  const bf16_t* Wb = W + (size_t)tn * 64 * K;

  v8f acc[2][4] = {};
  for (int k = 0; k < K; k += 32) {
    v16bf a0 = load_frag(Ab + k, lda);
    v16bf a1 = load_frag(Ab + (size_t)16 * lda + k, lda);
#pragma unroll
    for (int j = 0; j < 4; ++j) {
      v16bf b = load_frag(Wb + (size_t)j * 16 * K + k, K);
      acc[0][j] = wmma_bf16(a0, b, acc[0][j]);
      acc[1][j] = wmma_bf16(a1, b, acc[1][j]);
    }
  }
  // C/D layout: lane<16 col=lane rows 0..7; lane>=16 col=lane-16 rows 8..15
  const int col = lane & 15;
  const int i0  = (lane >> 4) << 3;
  const size_t n0 = (size_t)tn * 64;
#pragma unroll
  for (int hh = 0; hh < 2; ++hh) {
    float* Crow = Cmat + ((size_t)tm * 32 + hh * 16 + i0) * N + n0;
#pragma unroll
    for (int j = 0; j < 4; ++j) {
      const float bb = bias[n0 + j * 16 + col];
#pragma unroll
      for (int i = 0; i < 8; ++i)
        Crow[(size_t)i * N + j * 16 + col] = acc[hh][j][i] + bb;
    }
  }
}

// -------- GRU gate update (encoder step) -----------------------------------
__global__ __launch_bounds__(256) void gru_gate(
    const float* __restrict__ gi, const float* __restrict__ gh,
    float* __restrict__ h, bf16_t* __restrict__ h_bf) {
  const long long total = (long long)cfg::R * cfg::D;
  const long long idx = (long long)blockIdx.x * blockDim.x + threadIdx.x;
  if (idx >= total) return;
  const int d = (int)(idx % cfg::D);
  const long long r = idx / cfg::D;
  const float* gir = gi + (size_t)r * cfg::G;
  const float* ghr = gh + (size_t)r * cfg::G;
  const float rg = sigmoidf_(gir[d] + ghr[d]);
  const float z  = sigmoidf_(gir[d + cfg::D] + ghr[d + cfg::D]);
  const float n  = tanhf(gir[d + 2 * cfg::D] + rg * ghr[d + 2 * cfg::D]);
  const float hn = (1.f - z) * n + z * h[idx];
  h[idx] = hn;
  h_bf[idx] = (bf16_t)hn;
}

// -------- decoder input (depends only on (c, s); rows >= 4815 zero-padded) -
__global__ __launch_bounds__(256) void build_dec(
    const float* __restrict__ pos_emb, const float* __restrict__ channel_emb,
    bf16_t* __restrict__ dec_in) {
  const long long total = (long long)cfg::MP * cfg::D;
  const long long i = (long long)blockIdx.x * blockDim.x + threadIdx.x;
  if (i >= total) return;
  const int d = (int)(i % cfg::D);
  const long long row = i / cfg::D;
  float v = 0.f;
  if (row < (long long)cfg::C * cfg::S) {
    const int s = (int)(row % cfg::S);
    const int c = (int)(row / cfg::S);
    v = (d < cfg::D / 2) ? pos_emb[(size_t)s * (cfg::D / 2) + d]
                         : channel_emb[(size_t)c * (cfg::D / 2) + (d - cfg::D / 2)];
  }
  dec_in[i] = (bf16_t)v;
}

// -------- decoder GRU gate: gi broadcast over batch, gh broadcast over s ---
__global__ __launch_bounds__(256) void dec_gate(
    const float* __restrict__ gi_s, const float* __restrict__ ghb,
    const float* __restrict__ h, bf16_t* __restrict__ hy) {
  const long long total = cfg::RS * cfg::D;
  const long long idx = (long long)blockIdx.x * blockDim.x + threadIdx.x;
  if (idx >= total) return;
  const int d = (int)(idx % cfg::D);
  const long long rr = idx / cfg::D;
  const int s = (int)(rr % cfg::S);
  const long long r = rr / cfg::S;
  const int c = (int)(r % cfg::C);
  const float* gi = gi_s + ((size_t)(c * cfg::S + s)) * cfg::G;
  const float* gh = ghb + (size_t)r * cfg::G;
  const float rg = sigmoidf_(gi[d] + gh[d]);
  const float z  = sigmoidf_(gi[d + cfg::D] + gh[d + cfg::D]);
  const float n  = tanhf(gi[d + 2 * cfg::D] + rg * gh[d + 2 * cfg::D]);
  const float hn = (1.f - z) * n + z * h[(size_t)r * cfg::D + d];
  hy[idx] = (bf16_t)hn;
}

// -------- projection GEMM (N=48) fused with transpose + seq_last add -------
// wave tile 32(M) x 48(N): 2 A frags x 3 B frags -> 6 WMMA per K=32 step.
__global__ __launch_bounds__(256) void pred_gemm(
    const bf16_t* __restrict__ hy, const bf16_t* __restrict__ Wp,
    const float* __restrict__ bp, const float* __restrict__ x_enc,
    float* __restrict__ out) {
  const int wave = threadIdx.x >> 5;
  const int lane = threadIdx.x & 31;
  const long long tiles = cfg::RS / 32;  // 4815
  const long long tm = (long long)blockIdx.x * (blockDim.x >> 5) + wave;
  if (tm >= tiles) return;               // wave-uniform guard
  const bf16_t* Ab = hy + (size_t)tm * 32 * cfg::D;
  v8f acc[2][3] = {};
  for (int k = 0; k < cfg::D; k += 32) {
    v16bf a0 = load_frag(Ab + k, cfg::D);
    v16bf a1 = load_frag(Ab + (size_t)16 * cfg::D + k, cfg::D);
#pragma unroll
    for (int j = 0; j < 3; ++j) {
      v16bf b = load_frag(Wp + (size_t)j * 16 * cfg::D + k, cfg::D);
      acc[0][j] = wmma_bf16(a0, b, acc[0][j]);
      acc[1][j] = wmma_bf16(a1, b, acc[1][j]);
    }
  }
  const int col = lane & 15;
  const int i0  = (lane >> 4) << 3;
#pragma unroll
  for (int hh = 0; hh < 2; ++hh) {
#pragma unroll
    for (int j = 0; j < 3; ++j) {
      const int t = j * 16 + col;
      const float bb = bp[t];
#pragma unroll
      for (int i = 0; i < 8; ++i) {
        const long long rr = tm * 32 + hh * 16 + i0 + i;
        const int s = (int)(rr % cfg::S);
        const long long r = rr / cfg::S;
        const int b = (int)(r / cfg::C);
        const int c = (int)(r % cfg::C);
        const float last = x_enc[((size_t)b * cfg::L + (cfg::L - 1)) * cfg::C + c];
        out[((size_t)b * cfg::L + (s * cfg::SEG + t)) * cfg::C + c] =
            acc[hh][j][i] + bb + last;
      }
    }
  }
}

extern "C" void kernel_launch(void* const* d_in, const int* in_sizes, int n_in,
                              void* d_out, int out_size, void* d_ws, size_t ws_size,
                              hipStream_t stream) {
  (void)in_sizes; (void)n_in; (void)out_size; (void)ws_size;
  using namespace cfg;
  const float* x_enc       = (const float*)d_in[0];
  const float* W_emb       = (const float*)d_in[1];
  const float* b_emb       = (const float*)d_in[2];
  const float* W_ih        = (const float*)d_in[3];
  const float* W_hh        = (const float*)d_in[4];
  const float* b_ih        = (const float*)d_in[5];
  const float* b_hh        = (const float*)d_in[6];
  const float* pos_emb     = (const float*)d_in[7];
  const float* channel_emb = (const float*)d_in[8];
  const float* W_pred      = (const float*)d_in[9];
  const float* b_pred      = (const float*)d_in[10];
  float* out = (float*)d_out;

  // ---- carve workspace (all offsets 256B aligned) ----
  char* ws = (char*)d_ws;
  auto carve = [&](size_t bytes) -> void* {
    void* p = ws;
    ws += (bytes + 255) & ~(size_t)255;
    return p;
  };
  bf16_t* W_ih_bf   = (bf16_t*)carve((size_t)G * D * 2);
  bf16_t* W_hh_bf   = (bf16_t*)carve((size_t)G * D * 2);
  bf16_t* W_pred_bf = (bf16_t*)carve((size_t)SEG * D * 2);
  float*  h_f       = (float*) carve((size_t)R * D * 4);
  bf16_t* h_bf      = (bf16_t*)carve((size_t)R * D * 2);
  float*  gi        = (float*) carve((size_t)R * G * 4);
  float*  gh        = (float*) carve((size_t)R * G * 4);
  float*  gi_small  = (float*) carve((size_t)MP * G * 4);
  bf16_t* dec_in    = (bf16_t*)carve((size_t)MP * D * 2);
  bf16_t* X_emb     = (bf16_t*)carve((size_t)RS * D * 2);
  bf16_t* hy        = X_emb;  // X_emb dead after the encoder; same size — reuse

  const int TPB = 256;
  auto blocks1d = [](long long n) { return (unsigned)((n + 255) / 256); };
  auto gemm_blocks = [](int M, int N) {
    long long t = (long long)(M >> 5) * (N >> 6);
    return (unsigned)((t + 7) / 8);
  };

  // 1. weights -> bf16, h = 0
  prep_kernel<<<blocks1d((long long)R * D), TPB, 0, stream>>>(
      W_ih, W_hh, W_pred, W_ih_bf, W_hh_bf, W_pred_bf, h_f, h_bf);

  // 2. segment embedding
  embed_kernel<<<(unsigned)RS, TPB, 0, stream>>>(x_enc, W_emb, b_emb, X_emb);

  // 3. GRU encoder scan: per step two WMMA GEMMs + gate fusion
  for (int s = 0; s < S; ++s) {
    gemm_bf16_bias<<<gemm_blocks(R, G), TPB, 0, stream>>>(
        X_emb + (size_t)s * D, S * D, W_ih_bf, b_ih, gi, R, G, D);
    gemm_bf16_bias<<<gemm_blocks(R, G), TPB, 0, stream>>>(
        h_bf, D, W_hh_bf, b_hh, gh, R, G, D);
    gru_gate<<<blocks1d((long long)R * D), TPB, 0, stream>>>(gi, gh, h_f, h_bf);
  }

  // 4. decoder: dec_in has only C*S unique rows -> 32x fewer gi FLOPs
  build_dec<<<blocks1d((long long)MP * D), TPB, 0, stream>>>(pos_emb, channel_emb,
                                                             dec_in);
  gemm_bf16_bias<<<gemm_blocks(MP, G), TPB, 0, stream>>>(
      dec_in, D, W_ih_bf, b_ih, gi_small, MP, G, D);
  gemm_bf16_bias<<<gemm_blocks(R, G), TPB, 0, stream>>>(
      h_bf, D, W_hh_bf, b_hh, gh, R, G, D);  // gh of hn, broadcast over s
  dec_gate<<<blocks1d(RS * (long long)D), TPB, 0, stream>>>(gi_small, gh, h_f, hy);

  // 5. projection + transpose + seq_last re-add
  pred_gemm<<<(unsigned)((RS / 32 + 7) / 8), TPB, 0, stream>>>(hy, W_pred_bf,
                                                               b_pred, x_enc, out);
}
